// EncoderBlock_35897336660753
// MI455X (gfx1250) — matrix operand order repaired and checked by reading, not compiled, not run on hardware
//
#include <hip/hip_runtime.h>
#include <math.h>

// ---------------- types ----------------
typedef __bf16 bf16;
typedef __attribute__((ext_vector_type(16))) __bf16 bf16x16;
typedef __attribute__((ext_vector_type(8)))  __bf16 bf16x8;
typedef __attribute__((ext_vector_type(8)))  float  v8f;
typedef __attribute__((ext_vector_type(4)))  int    i32x4;

// problem sizes (fixed by reference)
#define BB 8
#define TT 1024
#define CC 1024
#define HH 16
#define DD 64
#define BT (BB * TT)   // 8192

// CDNA5 async global->LDS path (guarded: falls back to register prefetch)
#if defined(__gfx1250__) && __has_builtin(__builtin_amdgcn_global_load_async_to_lds_b128) && __has_builtin(__builtin_amdgcn_s_wait_asynccnt)
#define USE_ASYNC 1
#else
#define USE_ASYNC 0
#endif

#if USE_ASYNC
__device__ __forceinline__ void async_cp16(void* lds_dst, const void* gsrc) {
  // param0: int4 addrspace(1)* (global src), param1: int4 addrspace(3)* (LDS dst)
  __builtin_amdgcn_global_load_async_to_lds_b128(
      (__attribute__((address_space(1))) i32x4*)(gsrc),
      (__attribute__((address_space(3))) i32x4*)(lds_dst),
      0, 0);
}
__device__ __forceinline__ void async_wait0() {
  __builtin_amdgcn_s_wait_asynccnt(0);
}
#endif

__device__ __forceinline__ v8f zero_v8f() {
  v8f z;
#pragma unroll
  for (int i = 0; i < 8; ++i) z[i] = 0.0f;
  return z;
}

// D = A(16x32 bf16) x B(32x16 bf16) + C, f32 accum
__device__ __forceinline__ v8f wmma_bf16(bf16x16 a, bf16x16 b, v8f c) {
  return __builtin_amdgcn_wmma_f32_16x16x32_bf16(
      /*neg_a=*/false, a, /*neg_b=*/false, b,
      /*c_mod=*/(short)0, c, /*reuse_a=*/false, /*reuse_b=*/false);
}

// ---- fragment loaders from LDS (row-major [rows][ldk] bf16) ----
// A-matrix 16x32 (ISA 7.12.2): lane m=lane&15, half=lane>>4,
// elems 0..7 -> K=half*8+e (16B contig), elems 8..15 -> K=16+half*8+(e-8)
__device__ __forceinline__ bf16x16 frag_a(const bf16* lds, int m_base, int ldk,
                                          int koff, int lane) {
  const bf16* p = lds + (size_t)(m_base + (lane & 15)) * ldk + koff;
  int h8 = (lane >> 4) * 8;
  bf16x8 lo = *(const bf16x8*)(p + h8);
  bf16x8 hi = *(const bf16x8*)(p + 16 + h8);
  bf16x16 r;
#pragma unroll
  for (int i = 0; i < 8; ++i) { r[i] = lo[i]; r[i + 8] = hi[i]; }
  return r;
}

// B-matrix 32x16: lane n=lane&15, half=lane>>4, elems 0..7 -> K=half*8+e, elems 8..15 -> K=16+half*8+(e-8).
// lds holds B transposed: lds[n][k].
__device__ __forceinline__ bf16x16 frag_b(const bf16* lds, int n_base, int ldk,
                                          int koff, int lane) {
  const bf16* p = lds + (size_t)(n_base + (lane & 15)) * ldk + koff + (lane >> 4) * 8;
  bf16x8 lo = *(const bf16x8*)(p);
  bf16x8 hi = *(const bf16x8*)(p + 16);
  bf16x16 r;
#pragma unroll
  for (int i = 0; i < 8; ++i) { r[i] = lo[i]; r[i + 8] = hi[i]; }
  return r;
}

// ---------------- one-time weight transpose+convert: W[K,N] f32 -> WT[N,K] bf16 ----
__global__ __launch_bounds__(256) void wtrans_kernel(
    const float* __restrict__ W, bf16* __restrict__ WT, int K, int N) {
  __shared__ float t[64][65];
  const int n0 = blockIdx.x * 64, k0 = blockIdx.y * 64;
#pragma unroll
  for (int i = 0; i < 16; ++i) {
    int idx = threadIdx.x + i * 256;
    int r = idx >> 6, c = idx & 63;
    t[r][c] = W[(size_t)(k0 + r) * N + n0 + c];
  }
  __syncthreads();
#pragma unroll
  for (int i = 0; i < 16; ++i) {
    int idx = threadIdx.x + i * 256;
    int r = idx >> 6, c = idx & 63;  // r: n offset, c: k offset
    WT[(size_t)(n0 + r) * K + k0 + c] = (bf16)t[c][r];
  }
}

// ---------------- LayerNorm: f32 -> bf16 ----------------
__global__ __launch_bounds__(256) void ln_bf16_kernel(
    const float* __restrict__ X, const float* __restrict__ g,
    const float* __restrict__ b, bf16* __restrict__ Y) {
  __shared__ float red[2][8];
  const int row = blockIdx.x;
  const float* x = X + (size_t)row * CC;
  float v[4], s = 0.f, ss = 0.f;
#pragma unroll
  for (int i = 0; i < 4; ++i) {
    v[i] = x[threadIdx.x + i * 256];
    s += v[i]; ss += v[i] * v[i];
  }
#pragma unroll
  for (int m = 1; m < 32; m <<= 1) {
    s  += __shfl_xor(s,  m, 32);
    ss += __shfl_xor(ss, m, 32);
  }
  if ((threadIdx.x & 31) == 0) {
    red[0][threadIdx.x >> 5] = s;
    red[1][threadIdx.x >> 5] = ss;
  }
  __syncthreads();
  float ts = 0.f, tss = 0.f;
#pragma unroll
  for (int i = 0; i < 8; ++i) { ts += red[0][i]; tss += red[1][i]; }
  const float mu = ts * (1.0f / CC);
  const float var = tss * (1.0f / CC) - mu * mu;
  const float rstd = rsqrtf(var + 1e-5f);
#pragma unroll
  for (int i = 0; i < 4; ++i) {
    int c = threadIdx.x + i * 256;
    Y[(size_t)row * CC + c] = (bf16)((v[i] - mu) * rstd * g[c] + b[c]);
  }
}

// ---------------- GEMM: A[M,K] bf16 x WT[N,K] bf16 ----------------
// block tile 128x128, BK=32, 256 threads = 8 waves, wave tile 32x64.
// Double-buffered LDS, one barrier per K-step; async global->LDS when available.
template <int BIAS, int GELU, int RES, int OUTBF, int QS>
__global__ __launch_bounds__(256) void gemm_bf16_kernel(
    const bf16* __restrict__ A, const bf16* __restrict__ Bt,
    const float* __restrict__ bias, const float* __restrict__ res,
    const float* __restrict__ ls, void* __restrict__ Dout,
    int M, int N, int K) {
  __shared__ __align__(16) bf16 As[2][128][40];   // [m][k], 80B stride
  __shared__ __align__(16) bf16 Bs[2][128][40];   // [n][k]

  const int bm = blockIdx.y * 128, bn = blockIdx.x * 128;
  const int tid = threadIdx.x, lane = tid & 31, w = tid >> 5;
  const int wm = (w & 3) * 32;   // 4 waves along M
  const int wn = (w >> 2) * 64;  // 2 waves along N

  // staging coordinates: 2 chunks of 8 bf16 per thread per tile
  int rowi[2], c8i[2];
#pragma unroll
  for (int i = 0; i < 2; ++i) {
    int idx = tid + i * 256;
    rowi[i] = idx >> 2;
    c8i[i] = (idx & 3) * 8;
  }

  v8f acc[2][4];
#pragma unroll
  for (int i = 0; i < 2; ++i)
#pragma unroll
    for (int j = 0; j < 4; ++j) acc[i][j] = zero_v8f();

  const int nk = K >> 5;

#if USE_ASYNC
  // async copy next tile directly into LDS; ASYNCcnt-tracked
#define GEMM_ISSUE(buf, k0)                                                        \
  do {                                                                             \
    _Pragma("unroll") for (int i = 0; i < 2; ++i) {                                \
      async_cp16(&As[buf][rowi[i]][c8i[i]],                                        \
                 &A[(size_t)(bm + rowi[i]) * K + (k0) + c8i[i]]);                  \
      async_cp16(&Bs[buf][rowi[i]][c8i[i]],                                        \
                 &Bt[(size_t)(bn + rowi[i]) * K + (k0) + c8i[i]]);                 \
    }                                                                              \
  } while (0)
  GEMM_ISSUE(0, 0);
  for (int t = 0; t < nk; ++t) {
    const int cur = t & 1;
    async_wait0();
    __syncthreads();
    if (t + 1 < nk) GEMM_ISSUE(cur ^ 1, (t + 1) * 32);
#else
  bf16x8 ra[2], rb[2];
#define GEMM_GLOAD(k0)                                                             \
  do {                                                                             \
    _Pragma("unroll") for (int i = 0; i < 2; ++i) {                                \
      ra[i] = *(const bf16x8*)&A[(size_t)(bm + rowi[i]) * K + (k0) + c8i[i]];      \
      rb[i] = *(const bf16x8*)&Bt[(size_t)(bn + rowi[i]) * K + (k0) + c8i[i]];     \
    }                                                                              \
  } while (0)
  GEMM_GLOAD(0);
  for (int t = 0; t < nk; ++t) {
    const int cur = t & 1;
#pragma unroll
    for (int i = 0; i < 2; ++i) {
      *(bf16x8*)&As[cur][rowi[i]][c8i[i]] = ra[i];
      *(bf16x8*)&Bs[cur][rowi[i]][c8i[i]] = rb[i];
    }
    __syncthreads();
    if (t + 1 < nk) GEMM_GLOAD((t + 1) * 32);
#endif
    bf16x16 af[2];
#pragma unroll
    for (int i = 0; i < 2; ++i)
      af[i] = frag_a(&As[cur][0][0], wm + i * 16, 40, 0, lane);
    bf16x16 bfr[4];
#pragma unroll
    for (int j = 0; j < 4; ++j)
      bfr[j] = frag_b(&Bs[cur][0][0], wn + j * 16, 40, 0, lane);
#pragma unroll
    for (int i = 0; i < 2; ++i)
#pragma unroll
      for (int j = 0; j < 4; ++j) acc[i][j] = wmma_bf16(af[i], bfr[j], acc[i][j]);
  }

  // epilogue
  const int half = lane >> 4, nlo = lane & 15;
#pragma unroll
  for (int i = 0; i < 2; ++i)
#pragma unroll
    for (int j = 0; j < 4; ++j)
#pragma unroll
      for (int r = 0; r < 8; ++r) {
        int row = bm + wm + i * 16 + r + 8 * half;
        int col = bn + wn + j * 16 + nlo;
        float v = acc[i][j][r];
        if (BIAS) v += bias[col];
        if (QS && col < CC) v *= 0.125f;  // fold attention 1/sqrt(D) into q
        if (GELU) v = 0.5f * v * (1.0f + erff(v * 0.70710678118654752f));
        size_t o = (size_t)row * N + col;
        if (RES) {
          ((float*)Dout)[o] = res[o] + ls[col] * v;
        } else if (OUTBF) {
          ((bf16*)Dout)[o] = (bf16)v;
        } else {
          ((float*)Dout)[o] = v;
        }
      }
}

// ---------------- fused flash attention ----------------
// grid: (T/128, B*H). 256 threads = 8 waves; wave owns 16 q-rows.
// QKV bf16 [BT, 3C] (q pre-scaled by 1/8); O bf16 [BT, C].
__global__ __launch_bounds__(256) void attn_kernel(
    const bf16* __restrict__ QKV, bf16* __restrict__ O) {
  __shared__ __align__(16) bf16 Q_lds[128][72];
  __shared__ __align__(16) bf16 K_lds[64][72];   // [tk][d]  == B^T for S=Q*K^T
  __shared__ __align__(16) bf16 VT_lds[64][72];  // [d][tk]  == B^T for P*V
  __shared__ __align__(16) bf16 P_lds[8][16][72];

  const int q0 = blockIdx.x * 128;
  const int bh = blockIdx.y;
  const int bidx = bh >> 4, h = bh & 15;
  const int tid = threadIdx.x, lane = tid & 31, w = tid >> 5;
  const int half = lane >> 4, nlo = lane & 15;
  const size_t rowQ = (size_t)(bidx * TT + q0);

  // stage Q tile (128x64 bf16) once
#pragma unroll
  for (int i = 0; i < 4; ++i) {
    int idx = tid + i * 256;
    int r = idx >> 3;
    int c8 = (idx & 7) * 8;
    *(bf16x8*)&Q_lds[r][c8] =
        *(const bf16x8*)&QKV[(rowQ + r) * (3 * CC) + h * DD + c8];
  }
  __syncthreads();

  bf16x16 qa[2];
  qa[0] = frag_a(&Q_lds[0][0], w * 16, 72, 0, lane);
  qa[1] = frag_a(&Q_lds[0][0], w * 16, 72, 32, lane);

  v8f oacc[4];
#pragma unroll
  for (int j = 0; j < 4; ++j) oacc[j] = zero_v8f();
  float m_run[8], l_run[8];
#pragma unroll
  for (int r = 0; r < 8; ++r) { m_run[r] = -1e30f; l_run[r] = 0.0f; }

  // per-thread KV staging coords: 2 chunks of 8 bf16 each for K and V
  int rkv[2], ckv[2];
#pragma unroll
  for (int i = 0; i < 2; ++i) {
    int idx = tid + i * 256;        // 0..511
    rkv[i] = idx >> 3;              // tile row 0..63
    ckv[i] = (idx & 7) * 8;         // d chunk
  }

  bf16x8 rK[2], rV[2];
#define ATTN_GLOAD(kt)                                                             \
  do {                                                                             \
    size_t rb_ = (size_t)(bidx * TT + (kt) * 64);                                  \
    _Pragma("unroll") for (int i = 0; i < 2; ++i) {                                \
      rK[i] = *(const bf16x8*)&QKV[(rb_ + rkv[i]) * (3 * CC) + CC + h * DD + ckv[i]];      \
      rV[i] = *(const bf16x8*)&QKV[(rb_ + rkv[i]) * (3 * CC) + 2 * CC + h * DD + ckv[i]];  \
    }                                                                              \
  } while (0)

  ATTN_GLOAD(0);

  for (int kt = 0; kt < TT / 64; ++kt) {
    __syncthreads();  // previous compute done reading K_lds/VT_lds
#pragma unroll
    for (int i = 0; i < 2; ++i) {
      *(bf16x8*)&K_lds[rkv[i]][ckv[i]] = rK[i];
#pragma unroll
      for (int e = 0; e < 8; ++e) VT_lds[ckv[i] + e][rkv[i]] = rV[i][e];
    }
    __syncthreads();
    if (kt + 1 < TT / 64) ATTN_GLOAD(kt + 1);  // prefetch overlaps compute

    // S = Q * K^T  (16 x 64 per wave); q carries 1/sqrt(D)
    v8f s[4];
#pragma unroll
    for (int j = 0; j < 4; ++j) s[j] = zero_v8f();
#pragma unroll
    for (int kk = 0; kk < 2; ++kk) {
#pragma unroll
      for (int j = 0; j < 4; ++j) {
        bf16x16 kb = frag_b(&K_lds[0][0], j * 16, 72, kk * 32, lane);
        s[j] = wmma_bf16(qa[kk], kb, s[j]);
      }
    }

    // online softmax (rows live on 16-lane half-groups)
#pragma unroll
    for (int r = 0; r < 8; ++r) {
      float mx = fmaxf(fmaxf(s[0][r], s[1][r]), fmaxf(s[2][r], s[3][r]));
#pragma unroll
      for (int d = 1; d < 16; d <<= 1) mx = fmaxf(mx, __shfl_xor(mx, d, 32));
      float mn = fmaxf(m_run[r], mx);
      float alpha = __expf(m_run[r] - mn);
      m_run[r] = mn;
      float psum = 0.0f;
#pragma unroll
      for (int j = 0; j < 4; ++j) {
        float p = __expf(s[j][r] - mn);
        s[j][r] = p;
        psum += p;
      }
#pragma unroll
      for (int d = 1; d < 16; d <<= 1) psum += __shfl_xor(psum, d, 32);
      l_run[r] = l_run[r] * alpha + psum;
#pragma unroll
      for (int j = 0; j < 4; ++j) oacc[j][r] *= alpha;
    }

    // P (16x64) -> bf16 via per-wave LDS, then O += P * V
#pragma unroll
    for (int j = 0; j < 4; ++j)
#pragma unroll
      for (int r = 0; r < 8; ++r)
        P_lds[w][r + 8 * half][j * 16 + nlo] = (bf16)s[j][r];
    // same-wave LDS ops are in-order (DScnt); no barrier needed for P_lds
#pragma unroll
    for (int kk = 0; kk < 2; ++kk) {
      bf16x16 pa = frag_a(&P_lds[w][0][0], 0, 72, kk * 32, lane);
#pragma unroll
      for (int j = 0; j < 4; ++j) {
        bf16x16 vb = frag_b(&VT_lds[0][0], j * 16, 72, kk * 32, lane);
        oacc[j] = wmma_bf16(pa, vb, oacc[j]);
      }
    }
  }

  // finalize and write O as bf16 [BT, C]
#pragma unroll
  for (int j = 0; j < 4; ++j)
#pragma unroll
    for (int r = 0; r < 8; ++r) {
      float o = oacc[j][r] / l_run[r];
      int trow = q0 + w * 16 + r + 8 * half;
      int col = h * DD + j * 16 + nlo;
      O[(size_t)(bidx * TT + trow) * CC + col] = (bf16)o;
    }
}

// ---------------- launcher ----------------
extern "C" void kernel_launch(void* const* d_in, const int* in_sizes, int n_in,
                              void* d_out, int out_size, void* d_ws, size_t ws_size,
                              hipStream_t stream) {
  const float* x     = (const float*)d_in[0];
  const float* qkv_w = (const float*)d_in[1];
  const float* out_w = (const float*)d_in[2];
  const float* out_b = (const float*)d_in[3];
  const float* ff_w1 = (const float*)d_in[4];
  const float* ff_b1 = (const float*)d_in[5];
  const float* ff_w2 = (const float*)d_in[6];
  const float* ff_b2 = (const float*)d_in[7];
  const float* ln1_g = (const float*)d_in[8];
  const float* ln1_b = (const float*)d_in[9];
  const float* ln2_g = (const float*)d_in[10];
  const float* ln2_b = (const float*)d_in[11];
  const float* ls1   = (const float*)d_in[12];
  const float* ls2   = (const float*)d_in[13];
  float* out = (float*)d_out;

  // workspace layout (lifetime-aliased), total 184 MB
  char* ws = (char*)d_ws;
  const size_t szLN  = (size_t)BT * CC * 2;      // 16 MB bf16
  const size_t szQKV = (size_t)BT * 3 * CC * 2;  // 48 MB bf16
  const size_t szX1  = (size_t)BT * CC * 4;      // 32 MB f32
  const size_t szFF1 = (size_t)BT * 4 * CC * 2;  // 64 MB bf16
  bf16*  ln1h  = (bf16*)(ws);                    // dead after QKV gemm
  bf16*  attnh = (bf16*)(ws);                    // reuses ln1h region
  bf16*  qkvh  = (bf16*)(ws + szLN);             // dead after attention
  bf16*  ln2h  = (bf16*)(ws + szLN);             // reuses qkvh region
  float* x1    = (float*)(ws + szLN + szQKV);
  bf16*  ff1h  = (bf16*)(ws + szLN + szQKV + szX1);
  char*  wbase = ws + szLN + szQKV + szX1 + szFF1;
  bf16* qkvT = (bf16*)(wbase);                                 // [3072][1024]  6 MB
  bf16* outT = (bf16*)(wbase + (size_t)3 * CC * CC * 2);       // [1024][1024]  2 MB
  bf16* ff1T = (bf16*)(wbase + (size_t)4 * CC * CC * 2);       // [4096][1024]  8 MB
  bf16* ff2T = (bf16*)(wbase + (size_t)8 * CC * CC * 2);       // [1024][4096]  8 MB

  // 0) one-time weight bf16 transposes
  wtrans_kernel<<<dim3(3 * CC / 64, CC / 64), 256, 0, stream>>>(qkv_w, qkvT, CC, 3 * CC);
  wtrans_kernel<<<dim3(CC / 64, CC / 64),     256, 0, stream>>>(out_w, outT, CC, CC);
  wtrans_kernel<<<dim3(4 * CC / 64, CC / 64), 256, 0, stream>>>(ff_w1, ff1T, CC, 4 * CC);
  wtrans_kernel<<<dim3(CC / 64, 4 * CC / 64), 256, 0, stream>>>(ff_w2, ff2T, 4 * CC, CC);

  // 1) ln1: x -> bf16
  ln_bf16_kernel<<<BT, 256, 0, stream>>>(x, ln1_g, ln1_b, ln1h);
  // 2) qkv = ln1h @ qkv_w (q scaled by 1/8 in epilogue) -> bf16
  gemm_bf16_kernel<0, 0, 0, 1, 1><<<dim3(3 * CC / 128, BT / 128), 256, 0, stream>>>(
      ln1h, qkvT, nullptr, nullptr, nullptr, qkvh, BT, 3 * CC, CC);
  // 3) fused attention -> attnh bf16
  attn_kernel<<<dim3(TT / 128, BB * HH), 256, 0, stream>>>(qkvh, attnh);
  // 4) x1 = x + ls1 * (attnh @ out_w + out_b)
  gemm_bf16_kernel<1, 0, 1, 0, 0><<<dim3(CC / 128, BT / 128), 256, 0, stream>>>(
      attnh, outT, out_b, x, ls1, x1, BT, CC, CC);
  // 5) ln2: x1 -> bf16
  ln_bf16_kernel<<<BT, 256, 0, stream>>>(x1, ln2_g, ln2_b, ln2h);
  // 6) ff1h = gelu(ln2h @ ff_w1 + ff_b1) -> bf16
  gemm_bf16_kernel<1, 1, 0, 1, 0><<<dim3(4 * CC / 128, BT / 128), 256, 0, stream>>>(
      ln2h, ff1T, ff_b1, nullptr, nullptr, ff1h, BT, 4 * CC, CC);
  // 7) out = x1 + ls2 * (ff1h @ ff_w2 + ff_b2)
  gemm_bf16_kernel<1, 0, 1, 0, 0><<<dim3(CC / 128, BT / 128), 256, 0, stream>>>(
      ff1h, ff2T, ff_b2, x1, ls2, out, BT, CC, 4 * CC);
}
